// MultiHeadAttention_90632399880402
// MI455X (gfx1250) — matrix operand order repaired and checked by reading, not compiled, not run on hardware
//
#include <hip/hip_runtime.h>
#include <hip/hip_bf16.h>

typedef __bf16 bf16;
typedef __attribute__((ext_vector_type(16))) __bf16 v16bf;
typedef __attribute__((ext_vector_type(8)))  __bf16 v8bf;
typedef __attribute__((ext_vector_type(8)))  float  v8f;
typedef __attribute__((ext_vector_type(4)))  int    v4i;

#define WMMA_BF16F32(A,B,C) \
  __builtin_amdgcn_wmma_f32_16x16x32_bf16(false,(A),false,(B),(short)0,(C),false,false)

// ---- CDNA5 async global->LDS copy (ASYNCcnt-tracked).
#if defined(__has_builtin)
#  if __has_builtin(__builtin_amdgcn_global_load_async_to_lds_b128) && \
      __has_builtin(__builtin_amdgcn_s_wait_asynccnt)
#    define USE_ASYNC 1
#  endif
#endif
#ifndef USE_ASYNC
#  define USE_ASYNC 0
#endif

#if USE_ASYNC
typedef __attribute__((address_space(1))) v4i* gas_v4i;
typedef __attribute__((address_space(3))) v4i* las_v4i;
static __device__ __forceinline__ void async_cp16(const void* g, void* l) {
  __builtin_amdgcn_global_load_async_to_lds_b128((gas_v4i)g, (las_v4i)l, 0, 0);
}
static __device__ __forceinline__ void async_wait0() {
  __builtin_amdgcn_s_wait_asynccnt(0);
}
#else
static __device__ __forceinline__ void async_wait0() {}
#endif

static __device__ __forceinline__ v16bf cat16(v8bf lo, v8bf hi) {
  v16bf r;
#pragma unroll
  for (int i = 0; i < 8; ++i) { r[i] = lo[i]; r[i + 8] = hi[i]; }
  return r;
}

constexpr int DM   = 768;     // d_model
constexpr int SEQ  = 2048;    // sequence length
constexpr int BATCH= 2;
constexpr int H    = 12;
constexpr int ROWS = BATCH * SEQ;  // 4096
constexpr int LDT  = 72;      // padded LDS stride (bf16 elems) to avoid bank conflicts

// ---------------------------------------------------------------------------
// Weight prepass: Wt[n*768+k] = bf16(W[k*768+n]), tiled 64x64 through LDS so
// both the fp32 reads and the bf16 writes are coalesced b128 accesses.
// ---------------------------------------------------------------------------
__global__ __launch_bounds__(256) void wt_kernel(const float* __restrict__ W,
                                                 bf16* __restrict__ Wt) {
  __shared__ bf16 T[64 * LDT];
  const int k0 = blockIdx.y * 64;   // source row block (k)
  const int n0 = blockIdx.x * 64;   // source col block (n)
  const int tid = threadIdx.x;
  const int r  = tid >> 2;          // 0..63
  const int c  = (tid & 3) * 16;    // 0,16,32,48

  const float4* s = (const float4*)(W + (size_t)(k0 + r) * DM + n0 + c);
  float tmp[16];
  *(float4*)&tmp[0]  = s[0];
  *(float4*)&tmp[4]  = s[1];
  *(float4*)&tmp[8]  = s[2];
  *(float4*)&tmp[12] = s[3];
#pragma unroll
  for (int i = 0; i < 16; ++i) T[r * LDT + c + i] = (bf16)tmp[i];
  __syncthreads();

  v8bf o0, o1;
#pragma unroll
  for (int i = 0; i < 8; ++i) {
    o0[i] = T[(c + i) * LDT + r];
    o1[i] = T[(c + 8 + i) * LDT + r];
  }
  bf16* dst = Wt + (size_t)(n0 + r) * DM + k0 + c;
  *(v8bf*)dst       = o0;
  *(v8bf*)(dst + 8) = o1;
}

// ---------------------------------------------------------------------------
// bf16 WMMA GEMM:  Out[4096x768] = A[4096x768] * W[768x768] + bias
//   Wt pre-transposed bf16 [n][k].  Block tile 64(M) x 128(N), 8 waves (2x4),
//   32x32 per wave (2x2 WMMA tiles).  Double-buffered 64-wide K chunks:
//   chunk c+1's async DMA is in flight while chunk c's WMMAs execute.
// ---------------------------------------------------------------------------
template <bool A_F32, bool OUT_BF16>
__global__ __launch_bounds__(256) void gemm_kernel(const void* __restrict__ Av,
                                                   const bf16* __restrict__ Wt,
                                                   const float* __restrict__ bias,
                                                   void* __restrict__ Outv) {
  __shared__ bf16 Ab[2][64 * LDT];
  __shared__ bf16 Bb[2][128 * LDT];

  const int tid  = threadIdx.x;
  const int w    = tid >> 5;
  const int lane = tid & 31;
  const int half = lane >> 4;
  const int l    = lane & 15;

  const int m0 = blockIdx.y * 64;
  const int n0 = blockIdx.x * 128;
  const int waveM = (w >> 2) * 32;
  const int waveN = (w & 3) * 32;

  const float* Af = (const float*)Av;
  const bf16*  Ah = (const bf16*)Av;

  // staging coordinates (fixed per thread)
  const int arow = tid >> 2, akk = (tid & 3) * 16;   // A: 16 elems/thread
  const int bn   = tid >> 1, bkk = (tid & 1) * 32;   // B: 32 elems/thread

  auto stage = [&](int k0, int buf) {
    // --- B tile (128 n x 64 k): async DMA straight into LDS
    {
      const bf16* src = Wt + (size_t)(n0 + bn) * DM + k0 + bkk;
      bf16*       dst = &Bb[buf][bn * LDT + bkk];
#if USE_ASYNC
#pragma unroll
      for (int i = 0; i < 4; ++i) async_cp16(src + i * 8, dst + i * 8);
#else
      const v8bf* s = (const v8bf*)src;
      *(v8bf*)dst        = s[0];
      *(v8bf*)(dst + 8)  = s[1];
      *(v8bf*)(dst + 16) = s[2];
      *(v8bf*)(dst + 24) = s[3];
#endif
    }
    // --- A tile (64 rows x 64 k)
    {
      bf16* dst = &Ab[buf][arow * LDT + akk];
      if constexpr (A_F32) {
        // fp32 -> bf16 conversion through VGPRs (overlaps async B copies)
        const float4* s = (const float4*)(Af + (size_t)(m0 + arow) * DM + k0 + akk);
        float tmp[16];
        *(float4*)&tmp[0]  = s[0];
        *(float4*)&tmp[4]  = s[1];
        *(float4*)&tmp[8]  = s[2];
        *(float4*)&tmp[12] = s[3];
        v8bf lo, hi;
#pragma unroll
        for (int i = 0; i < 8; ++i) { lo[i] = (bf16)tmp[i]; hi[i] = (bf16)tmp[8 + i]; }
        *(v8bf*)dst       = lo;
        *(v8bf*)(dst + 8) = hi;
        if (k0 + 64 < DM) __builtin_prefetch((const void*)(s + 4), 0, 0);
      } else {
        const bf16* src = Ah + (size_t)(m0 + arow) * DM + k0 + akk;
#if USE_ASYNC
        async_cp16(src, dst);
        async_cp16(src + 8, dst + 8);
#else
        const v8bf* s = (const v8bf*)src;
        *(v8bf*)dst       = s[0];
        *(v8bf*)(dst + 8) = s[1];
#endif
      }
    }
  };

  v8f acc[2][2];
#pragma unroll
  for (int i = 0; i < 2; ++i)
#pragma unroll
    for (int j = 0; j < 2; ++j) acc[i][j] = {};

  constexpr int NCHUNK = DM / 64;  // 12
  stage(0, 0);
  for (int c = 0; c < NCHUNK; ++c) {
    const int cur = c & 1;
    async_wait0();        // own chunk-c copies complete (per-wave counter)
    __syncthreads();      // everyone's chunk-c data published
    if (c + 1 < NCHUNK) stage((c + 1) * 64, cur ^ 1);  // DMA overlaps WMMAs

#pragma unroll
    for (int ks = 0; ks < 2; ++ks) {
      const int koff = ks * 32;
      v16bf afrag[2], bfrag[2];
#pragma unroll
      for (int mi = 0; mi < 2; ++mi) {
        // A-layout: lane half interleaves K by 8
        const bf16* p = &Ab[cur][(waveM + mi * 16 + l) * LDT + koff + half * 8];
        afrag[mi] = cat16(*(const v8bf*)p, *(const v8bf*)(p + 16));
      }
#pragma unroll
      for (int ni = 0; ni < 2; ++ni) {
        // B-layout: lane = column N, lane half interleaves K by 8 (matches A)
        const bf16* p = &Bb[cur][(waveN + ni * 16 + l) * LDT + koff + half * 8];
        bfrag[ni] = cat16(*(const v8bf*)p, *(const v8bf*)(p + 16));
      }
#pragma unroll
      for (int mi = 0; mi < 2; ++mi)
#pragma unroll
        for (int ni = 0; ni < 2; ++ni)
          acc[mi][ni] = WMMA_BF16F32(afrag[mi], bfrag[ni], acc[mi][ni]);
    }
    __syncthreads();      // compute(c) done before buffer is re-staged
  }

  // --- epilogue: C layout lane(half,l): VGPR r -> M = half*8+r, N = l
#pragma unroll
  for (int mi = 0; mi < 2; ++mi)
#pragma unroll
    for (int ni = 0; ni < 2; ++ni)
#pragma unroll
      for (int r = 0; r < 8; ++r) {
        int row = m0 + waveM + mi * 16 + half * 8 + r;
        int col = n0 + waveN + ni * 16 + l;
        float v = acc[mi][ni][r] + bias[col];
        if constexpr (OUT_BF16)
          ((bf16*)Outv)[(size_t)row * DM + col] = (bf16)v;
        else
          ((float*)Outv)[(size_t)row * DM + col] = v;
      }
}

// ---------------------------------------------------------------------------
// Flash attention: per block (b, h, 128 q-rows); 8 waves x 16 q-rows.
// Online softmax; 64-key chunks double-buffered: K via async DMA, V transposed
// through VGPRs (must land [d][key] for the PV WMMA B-fragments).
// ---------------------------------------------------------------------------
__global__ __launch_bounds__(256) void attn_kernel(const bf16* __restrict__ Q,
                                                   const bf16* __restrict__ K,
                                                   const bf16* __restrict__ V,
                                                   const int*  __restrict__ mask,
                                                   bf16* __restrict__ Ctx) {
  __shared__ bf16 KS[2][64 * LDT];         // [key][d]   -> B-frag for Q*K^T
  __shared__ bf16 VS[2][64 * LDT];         // [d][key]   -> B-frag for P*V
  __shared__ bf16 PS[8 * 16 * LDT];        // per-wave P tile [m][key] (A-layout rows)
  __shared__ int  MK[2][64];

  const int tid  = threadIdx.x;
  const int w    = tid >> 5;
  const int lane = tid & 31;
  const int half = lane >> 4;
  const int l    = lane & 15;

  const int q0 = blockIdx.x * 128;
  const int b  = blockIdx.y / H;
  const int h  = blockIdx.y % H;

  const bf16* Qg = Q + (size_t)b * SEQ * DM + h * 64;
  const bf16* Kg = K + (size_t)b * SEQ * DM + h * 64;
  const bf16* Vg = V + (size_t)b * SEQ * DM + h * 64;

  const int ldkey = tid >> 2;          // staging row (key) per thread
  const int lddd  = (tid & 3) * 16;    // staging d offset per thread

  auto stage = [&](int kc, int buf) {
    const int key0 = kc * 64;
    const bf16* ksrc = Kg + (size_t)(key0 + ldkey) * DM + lddd;
#if USE_ASYNC
    async_cp16(ksrc,     &KS[buf][ldkey * LDT + lddd]);
    async_cp16(ksrc + 8, &KS[buf][ldkey * LDT + lddd + 8]);
#else
    const v8bf* ks_ = (const v8bf*)ksrc;
    *(v8bf*)&KS[buf][ldkey * LDT + lddd]     = ks_[0];
    *(v8bf*)&KS[buf][ldkey * LDT + lddd + 8] = ks_[1];
#endif
    const bf16* vsrc = Vg + (size_t)(key0 + ldkey) * DM + lddd;
    const v8bf* vs_ = (const v8bf*)vsrc;
    v8bf v0 = vs_[0], v1 = vs_[1];
#pragma unroll
    for (int i = 0; i < 8; ++i) {
      VS[buf][(lddd + i) * LDT + ldkey]     = v0[i];
      VS[buf][(lddd + 8 + i) * LDT + ldkey] = v1[i];
    }
    if (tid < 64) MK[buf][tid] = mask[(size_t)b * SEQ + key0 + tid];
    if (kc + 1 < SEQ / 64) __builtin_prefetch(vsrc + (size_t)DM * 64, 0, 0);
  };

  // Q fragments (16 rows x 64 d = two 16x32 A-fragments), loaded once from HBM
  v16bf aq[2];
  {
    const bf16* qrow = Qg + (size_t)(q0 + w * 16 + l) * DM;
#pragma unroll
    for (int ks = 0; ks < 2; ++ks) {
      v8bf lo = *(const v8bf*)(qrow + ks * 32 + half * 8);
      v8bf hi = *(const v8bf*)(qrow + ks * 32 + 16 + half * 8);
      aq[ks] = cat16(lo, hi);
    }
  }

  float m_r[8], l_r[8];
  v8f cacc[4];
#pragma unroll
  for (int r = 0; r < 8; ++r) { m_r[r] = -1e30f; l_r[r] = 0.f; }
#pragma unroll
  for (int ni = 0; ni < 4; ++ni) cacc[ni] = {};

  constexpr int NCHUNK = SEQ / 64;  // 32
  stage(0, 0);
  for (int kc = 0; kc < NCHUNK; ++kc) {
    const int cur = kc & 1;
    async_wait0();        // own chunk-kc K copies complete
    __syncthreads();      // everyone's chunk-kc K/V/mask published
    if (kc + 1 < NCHUNK) stage(kc + 1, cur ^ 1);  // DMA overlaps score/softmax

    // --- scores = Q * K^T  (16 x 64), four 16x16 tiles
    v8f sacc[4];
#pragma unroll
    for (int ni = 0; ni < 4; ++ni) sacc[ni] = {};
#pragma unroll
    for (int ks = 0; ks < 2; ++ks) {
      const int koff = ks * 32;
#pragma unroll
      for (int ni = 0; ni < 4; ++ni) {
        const bf16* p = &KS[cur][(ni * 16 + l) * LDT + koff + half * 8];
        v16bf bfrag = cat16(*(const v8bf*)p, *(const v8bf*)(p + 16));
        sacc[ni] = WMMA_BF16F32(aq[ks], bfrag, sacc[ni]);
      }
    }

    // --- online softmax (rows M = half*8 + r live on the 16 lanes of a half)
    float sval[4][8];
    float mloc[8];
#pragma unroll
    for (int r = 0; r < 8; ++r) mloc[r] = -1e30f;
#pragma unroll
    for (int ni = 0; ni < 4; ++ni) {
      float keep = (MK[cur][ni * 16 + l] == 0) ? -1e30f : 0.f;
#pragma unroll
      for (int r = 0; r < 8; ++r) {
        float s = sacc[ni][r] * 0.125f + keep;   // scale 1/sqrt(64); mask -> -1e30
        sval[ni][r] = s;
        mloc[r] = fmaxf(mloc[r], s);
      }
    }
#pragma unroll
    for (int off = 1; off < 16; off <<= 1)
#pragma unroll
      for (int r = 0; r < 8; ++r)
        mloc[r] = fmaxf(mloc[r], __shfl_xor(mloc[r], off, 32));

    float alpha[8], rsum[8];
#pragma unroll
    for (int r = 0; r < 8; ++r) {
      float mn = fmaxf(m_r[r], mloc[r]);
      alpha[r] = __expf(m_r[r] - mn);
      m_r[r]   = mn;
      rsum[r]  = 0.f;
    }
#pragma unroll
    for (int ni = 0; ni < 4; ++ni)
#pragma unroll
      for (int r = 0; r < 8; ++r) {
        float p = __expf(sval[ni][r] - m_r[r]);
        rsum[r] += p;
        PS[(w * 16 + half * 8 + r) * LDT + ni * 16 + l] = (bf16)p;
      }
#pragma unroll
    for (int off = 1; off < 16; off <<= 1)
#pragma unroll
      for (int r = 0; r < 8; ++r)
        rsum[r] += __shfl_xor(rsum[r], off, 32);
#pragma unroll
    for (int r = 0; r < 8; ++r) l_r[r] = l_r[r] * alpha[r] + rsum[r];
#pragma unroll
    for (int ni = 0; ni < 4; ++ni)
#pragma unroll
      for (int r = 0; r < 8; ++r) cacc[ni][r] *= alpha[r];

    __syncthreads();   // P visible (cross-lane within wave; cheap block barrier)

    // --- ctx += P * V  (K-dim = keys, B = V^T staged in VS)
#pragma unroll
    for (int ks = 0; ks < 2; ++ks) {
      const int koff = ks * 32;
      const bf16* pa = &PS[(w * 16 + l) * LDT + koff + half * 8];
      v16bf ap = cat16(*(const v8bf*)pa, *(const v8bf*)(pa + 16));
#pragma unroll
      for (int ni = 0; ni < 4; ++ni) {
        const bf16* p = &VS[cur][(ni * 16 + l) * LDT + koff + half * 8];
        v16bf bfrag = cat16(*(const v8bf*)p, *(const v8bf*)(p + 16));
        cacc[ni] = WMMA_BF16F32(ap, bfrag, cacc[ni]);
      }
    }
    __syncthreads();   // compute(kc) done before its buffer is re-staged
  }

  // --- normalize and write ctx (bf16, [B,S,H*64] layout)
#pragma unroll
  for (int r = 0; r < 8; ++r) l_r[r] = 1.f / l_r[r];
#pragma unroll
  for (int ni = 0; ni < 4; ++ni)
#pragma unroll
    for (int r = 0; r < 8; ++r) {
      int row = q0 + w * 16 + half * 8 + r;
      int col = h * 64 + ni * 16 + l;
      Ctx[((size_t)b * SEQ + row) * DM + col] = (bf16)(cacc[ni][r] * l_r[r]);
    }
}

// ---------------------------------------------------------------------------
// Host-side orchestration
// ---------------------------------------------------------------------------
extern "C" void kernel_launch(void* const* d_in, const int* in_sizes, int n_in,
                              void* d_out, int out_size, void* d_ws, size_t ws_size,
                              hipStream_t stream) {
  const float* q    = (const float*)d_in[0];
  const float* k    = (const float*)d_in[1];
  const float* v    = (const float*)d_in[2];
  const int*   mask = (const int*)d_in[3];
  const float* Wq = (const float*)d_in[4];  const float* bq = (const float*)d_in[5];
  const float* Wk = (const float*)d_in[6];  const float* bk = (const float*)d_in[7];
  const float* Wv = (const float*)d_in[8];  const float* bv = (const float*)d_in[9];
  const float* Wo = (const float*)d_in[10]; const float* bo = (const float*)d_in[11];
  float* out = (float*)d_out;

  // workspace layout (bf16): 4 transposed weights + Q/K/V/Ctx activations (~30 MB)
  char* ws = (char*)d_ws;
  const size_t wsz = (size_t)DM * DM;        // weight elems
  const size_t asz = (size_t)ROWS * DM;      // activation elems
  bf16* WtQ = (bf16*)ws; ws += wsz * 2;
  bf16* WtK = (bf16*)ws; ws += wsz * 2;
  bf16* WtV = (bf16*)ws; ws += wsz * 2;
  bf16* WtO = (bf16*)ws; ws += wsz * 2;
  bf16* Qb  = (bf16*)ws; ws += asz * 2;
  bf16* Kb  = (bf16*)ws; ws += asz * 2;
  bf16* Vb  = (bf16*)ws; ws += asz * 2;
  bf16* Cb  = (bf16*)ws; ws += asz * 2;

  dim3 blk(256);
  dim3 wtg(DM / 64, DM / 64);     // (12, 12)
  wt_kernel<<<wtg, blk, 0, stream>>>(Wq, WtQ);
  wt_kernel<<<wtg, blk, 0, stream>>>(Wk, WtK);
  wt_kernel<<<wtg, blk, 0, stream>>>(Wv, WtV);
  wt_kernel<<<wtg, blk, 0, stream>>>(Wo, WtO);

  dim3 gg(DM / 128, ROWS / 64);   // (6, 64)
  gemm_kernel<true,  true ><<<gg, blk, 0, stream>>>(q, WtQ, bq, Qb);
  gemm_kernel<true,  true ><<<gg, blk, 0, stream>>>(k, WtK, bk, Kb);
  gemm_kernel<true,  true ><<<gg, blk, 0, stream>>>(v, WtV, bv, Vb);

  dim3 ga(SEQ / 128, BATCH * H);  // (16, 24)
  attn_kernel<<<ga, blk, 0, stream>>>(Qb, Kb, Vb, mask, Cb);

  gemm_kernel<false, false><<<gg, blk, 0, stream>>>(Cb, WtO, bo, out);
}
